// NonLocalBlock_43662637531286
// MI455X (gfx1250) — matrix operand order repaired and checked by reading, not compiled, not run on hardware
//
#include <hip/hip_runtime.h>

// ---------------------------------------------------------------------------
// NonLocalBlock for MI455X (gfx1250): bf16 WMMA GEMMs + flash attention,
// TDM (tensor_load_to_lds) for the attention Q tile, prefetch for K/V stream.
// b=8, c=512, hw=4096, groups=32.
// ---------------------------------------------------------------------------

typedef __attribute__((ext_vector_type(16))) __bf16 v16bf;
typedef __attribute__((ext_vector_type(8)))  float  v8f;
typedef __attribute__((ext_vector_type(4)))  unsigned int v4u;
typedef __attribute__((ext_vector_type(8)))  int    v8i;
typedef __attribute__((ext_vector_type(4)))  int    v4i;

#define B_   8
#define C_   512
#define S_   4096
#define GRP  32
#define CPG  16          // channels per group

__device__ __forceinline__ v8f wmma_bf16(v16bf a, v16bf b, v8f c) {
  return __builtin_amdgcn_wmma_f32_16x16x32_bf16(false, a, false, b, (short)0, c,
                                                 false, false);
}

// Build a 16-element bf16 fragment from two 16-byte chunks (LDS or global).
__device__ __forceinline__ v16bf make_frag2(const __bf16* lo, const __bf16* hi) {
  union { uint4 u[2]; v16bf v; } f;
  f.u[0] = *(const uint4*)lo;
  f.u[1] = *(const uint4*)hi;
  return f.v;
}
// 16 contiguous bf16 (B-fragments).
__device__ __forceinline__ v16bf make_frag16(const __bf16* p) {
  return make_frag2(p, p + 8);
}

// Wave-relative LDS byte offset of a shared-memory object (generic -> AS3).
__device__ __forceinline__ unsigned lds_offset_of(const void* p) {
  return (unsigned)(unsigned long long)
         (__attribute__((address_space(3))) const char*)p;
}

// ---------------------------------------------------------------------------
// 1) GroupNorm statistics: one block per (b, g). Group slab is contiguous.
// ---------------------------------------------------------------------------
__global__ __launch_bounds__(256, 1)
void gn_stats_kernel(const float* __restrict__ x, float* __restrict__ stats) {
  const int bg = blockIdx.x;                       // b*32 + g
  const float* base = x + (size_t)bg * CPG * S_;   // 16*4096 = 65536 elems
  float s = 0.f, ss = 0.f;
  for (int i = threadIdx.x; i < CPG * S_; i += 256) {
    float v = base[i];
    s += v; ss += v * v;
  }
  for (int m = 16; m >= 1; m >>= 1) {
    s  += __shfl_xor(s,  m, 32);
    ss += __shfl_xor(ss, m, 32);
  }
  __shared__ float rs[8], rss[8];
  if ((threadIdx.x & 31) == 0) { rs[threadIdx.x >> 5] = s; rss[threadIdx.x >> 5] = ss; }
  __syncthreads();
  if (threadIdx.x == 0) {
    float S = 0.f, SS = 0.f;
    for (int i = 0; i < 8; ++i) { S += rs[i]; SS += rss[i]; }
    const float inv_n = 1.0f / (float)(CPG * S_);
    float mu  = S * inv_n;
    float var = SS * inv_n - mu * mu;
    stats[bg * 2 + 0] = mu;
    stats[bg * 2 + 1] = rsqrtf(var + 1e-5f);
  }
}

// ---------------------------------------------------------------------------
// 2) Weights f32 -> bf16 (done once, then L2 resident)
// ---------------------------------------------------------------------------
__global__ __launch_bounds__(256, 1)
void wconv_kernel(const float* wq, const float* wk, const float* wv, const float* wo,
                  __bf16* q, __bf16* k, __bf16* v, __bf16* o) {
  const int n = C_ * C_;
  for (int i = blockIdx.x * 256 + threadIdx.x; i < n; i += gridDim.x * 256) {
    q[i] = (__bf16)wq[i];
    k[i] = (__bf16)wk[i];
    v[i] = (__bf16)wv[i];
    o[i] = (__bf16)wo[i];
  }
}

// ---------------------------------------------------------------------------
// 3) Fused GroupNorm-apply + Q/K/V projections.
//    Block tile: 64 spatial rows x 32 out-channels, K-loop over 512 in chunks
//    of 32. One LDS hidden tile feeds Q/K (as A) and V (as B, transposed out).
//    Q,K stored [b][s][c] bf16; V stored [b][c][s] bf16.
// ---------------------------------------------------------------------------
__global__ __launch_bounds__(256, 1)
void qkv_kernel(const float* __restrict__ x, const float* __restrict__ gamma,
                const float* __restrict__ beta, const float* __restrict__ stats,
                const __bf16* __restrict__ wqb, const __bf16* __restrict__ wkb,
                const __bf16* __restrict__ wvb,
                const float* __restrict__ bq, const float* __restrict__ bk,
                const float* __restrict__ bv,
                __bf16* __restrict__ Qg, __bf16* __restrict__ Kg,
                __bf16* __restrict__ Vt) {
  __shared__ __align__(16) __bf16 Hs[64][40];      // 64 s-rows x 32 ch (+pad)

  const int b  = blockIdx.z;
  const int s0 = blockIdx.x * 64;
  const int c0 = blockIdx.y * 32;                  // out-channel tile base
  const int tid  = threadIdx.x;
  const int w    = tid >> 5;
  const int lane = tid & 31;
  const int l16  = lane & 15;
  const int half = lane >> 4;

  // Q/K wave map: 4(M=s) x 2(N=cout); V wave map: 2(M=cout) x 4(N=s)
  const int mQ = 16 * (w >> 1), nQ = 16 * (w & 1);
  const int mV = 16 * (w >> 2), nV = 16 * (w & 3);

  v8f qa, ka, va;
  for (int e = 0; e < 8; ++e) { qa[e] = 0.f; ka[e] = 0.f; va[e] = 0.f; }

  const int ci_loc = tid >> 3;                     // 0..31 (channel within chunk)
  const int soff   = (tid & 7) * 8;                // 0..56

  for (int kc = 0; kc < C_; kc += 32) {
    __syncthreads();
    {  // load 64x32 x-tile, apply GroupNorm, store bf16 to LDS (transposed)
      const int ci = kc + ci_loc;
      const int g  = ci >> 4;
      const float mu   = stats[(b * GRP + g) * 2 + 0];
      const float rstd = stats[(b * GRP + g) * 2 + 1];
      const float ga = gamma[ci] * rstd;
      const float be = beta[ci] - mu * ga;
      const float* xs = x + ((size_t)b * C_ + ci) * S_ + s0 + soff;
      float4 v0 = *(const float4*)xs;
      float4 v1 = *(const float4*)(xs + 4);
      Hs[soff + 0][ci_loc] = (__bf16)(v0.x * ga + be);
      Hs[soff + 1][ci_loc] = (__bf16)(v0.y * ga + be);
      Hs[soff + 2][ci_loc] = (__bf16)(v0.z * ga + be);
      Hs[soff + 3][ci_loc] = (__bf16)(v0.w * ga + be);
      Hs[soff + 4][ci_loc] = (__bf16)(v1.x * ga + be);
      Hs[soff + 5][ci_loc] = (__bf16)(v1.y * ga + be);
      Hs[soff + 6][ci_loc] = (__bf16)(v1.z * ga + be);
      Hs[soff + 7][ci_loc] = (__bf16)(v1.w * ga + be);
    }
    __syncthreads();

    // Q/K: A = hidden rows (LDS), B = weight rows (global, L2 resident)
    v16bf a = make_frag2(&Hs[mQ + l16][half * 16], &Hs[mQ + l16][half * 16 + 8]);
    qa = wmma_bf16(a, make_frag16(wqb + (size_t)(c0 + nQ + l16) * C_ + kc + half * 16), qa);
    ka = wmma_bf16(a, make_frag16(wkb + (size_t)(c0 + nQ + l16) * C_ + kc + half * 16), ka);

    // V (transposed out): A = Wv rows (global), B = hidden rows (LDS)
    const __bf16* wvrow = wvb + (size_t)(c0 + mV + l16) * C_ + kc;
    v16bf av = make_frag2(&wvrow[half * 16], &wvrow[half * 16 + 8]);
    v16bf bvf = make_frag16(&Hs[nV + l16][half * 16]);
    va = wmma_bf16(av, bvf, va);
  }

  // store Q, K: [b][s][c]
  {
    const int cq = c0 + nQ + l16;
    const float biq = bq[cq], bik = bk[cq];
    for (int e = 0; e < 8; ++e) {
      const size_t idx = ((size_t)b * S_ + s0 + mQ + e + 8 * half) * C_ + cq;
      Qg[idx] = (__bf16)(qa[e] + biq);
      Kg[idx] = (__bf16)(ka[e] + bik);
    }
  }
  // store V: [b][c][s]
  for (int e = 0; e < 8; ++e) {
    const int cv = c0 + mV + e + 8 * half;
    Vt[((size_t)b * C_ + cv) * S_ + s0 + nV + l16] = (__bf16)(va[e] + bv[cv]);
  }
}

// ---------------------------------------------------------------------------
// 4) Flash attention. One block per (b, 32-row q tile). K/V stream from L2.
//    Q tile (32x512 bf16, padded rows) fetched by the Tensor Data Mover.
// ---------------------------------------------------------------------------
__global__ __launch_bounds__(256, 1)
void attn_kernel(const __bf16* __restrict__ Qg, const __bf16* __restrict__ Kg,
                 const __bf16* __restrict__ Vt, __bf16* __restrict__ Ah) {
  __shared__ __align__(16) __bf16 Qs[32][C_ + 8];
  __shared__ __align__(16) __bf16 Ps[32][72];
  __shared__ float redmax[4][32];
  __shared__ float redsum[4][32];
  __shared__ float rowm[32], rowl[32], rsf[32];

  const int b  = blockIdx.y;
  const int q0 = blockIdx.x * 32;
  const int tid  = threadIdx.x;
  const int w    = tid >> 5;
  const int lane = tid & 31;
  const int l16  = lane & 15;
  const int half = lane >> 4;

  const __bf16* Qb = Qg + (size_t)b * S_ * C_;
  const __bf16* Kb = Kg + (size_t)b * S_ * C_;
  const __bf16* Vb = Vt + (size_t)b * C_ * S_;     // [c][s]

#if __has_builtin(__builtin_amdgcn_tensor_load_to_lds) && \
    __has_builtin(__builtin_amdgcn_s_wait_tensorcnt)
  // --- TDM: one 2-D descriptor moves the whole 32x512 bf16 tile into LDS,
  //     inserting 4 DWORDs of pad every 256 DWORDs (row stride 520 elems).
  //     (6-arg builtin variant: g0, g1, g2, g3, extra group, cpol)
  if (w == 0) {
    const unsigned long long ga = (unsigned long long)(Qb + (size_t)q0 * C_);
    const unsigned lds = lds_offset_of(&Qs[0][0]);
    v4u g0;
    g0[0] = 1u;                                    // count=1, user descriptor
    g0[1] = lds;                                   // lds_addr
    g0[2] = (unsigned)(ga & 0xffffffffu);          // global_addr[31:0]
    g0[3] = (unsigned)((ga >> 32) & 0x1ffffffu)    // global_addr[56:32]
            | (2u << 30);                          // type = 2 ("image")
    v8i g1;
    g1[0] = (int)((1u << 16)                       // data_size = 2 bytes
                | (1u << 20)                       // pad_enable
                | (7u << 22)                       // pad_interval = 256 DWORDs
                | (3u << 25));                     // pad_amount  = 4 DWORDs
    g1[1] = (int)(512u << 16);                     // tensor_dim0 = 512
    g1[2] = (int)(4096u << 16);                    // tensor_dim1 = 4096
    g1[3] = (int)(512u << 16);                     // tile_dim0 = 512
    g1[4] = 32;                                    // tile_dim1 = 32 rows
    g1[5] = 512;                                   // tensor_dim0_stride = 512
    g1[6] = 0;                                     // dim1_stride[15:0] = 0
    g1[7] = 0x20;                                  // dim1_stride = 512*4096
    v4i z4; z4[0] = 0; z4[1] = 0; z4[2] = 0; z4[3] = 0;
    v8i z8; for (int e = 0; e < 8; ++e) z8[e] = 0;
    __builtin_amdgcn_tensor_load_to_lds(g0, g1, z4, z4, z8, 0);
    __builtin_amdgcn_s_wait_tensorcnt(0);
  }
#else
  // Fallback: cooperative Q tile load (32 x 512 bf16, 16B chunks)
  for (int i = tid; i < 32 * (C_ / 8); i += 256) {
    const int r  = i >> 6;
    const int c8 = (i & 63) << 3;
    *(uint4*)&Qs[r][c8] = *(const uint4*)&Qb[(size_t)(q0 + r) * C_ + c8];
  }
#endif
  if (tid < 32) { rowm[tid] = -3.0e38f; rowl[tid] = 0.f; }

  v8f O[2][4];
  for (int mi = 0; mi < 2; ++mi)
    for (int ni = 0; ni < 4; ++ni)
      for (int e = 0; e < 8; ++e) O[mi][ni][e] = 0.f;

  const float scale = 22.62741699796952f;          // sqrt(512), faithful to ref
  const int mS  = (w & 1) * 16;
  const int nQw = w >> 1;                          // 0..3
  const int nS  = nQw * 16;
  __syncthreads();

  for (int kt = 0; kt < S_ / 64; ++kt) {
    const int k0 = kt * 64;

    // prefetch next k-tile's K/V rows into cache (global_prefetch_b8)
    if (kt + 1 < S_ / 64) {
      __builtin_prefetch(Kb + (size_t)(k0 + 64 + nS + l16) * C_, 0, 0);
      __builtin_prefetch(Vb + (size_t)(64 * w + l16) * S_ + k0 + 64, 0, 0);
    }

    // ---- S = (Q K^T) tile, K-dim 512 in chunks of 32 ----
    v8f sacc;
    for (int e = 0; e < 8; ++e) sacc[e] = 0.f;
    const __bf16* Krow = Kb + (size_t)(k0 + nS + l16) * C_;
#pragma unroll 4
    for (int kc = 0; kc < C_; kc += 32) {
      v16bf a  = make_frag2(&Qs[mS + l16][kc + half * 16],
                            &Qs[mS + l16][kc + half * 16 + 8]);
      v16bf bb = make_frag16(&Krow[kc + half * 16]);
      sacc = wmma_bf16(a, bb, sacc);
    }

    // ---- per-row max within the tile (reduce across 16 lanes of each half)
    float rmax[8];
#pragma unroll
    for (int e = 0; e < 8; ++e) {
      float v = sacc[e] * scale;
      sacc[e] = v;
      for (int m = 8; m >= 1; m >>= 1) v = fmaxf(v, __shfl_xor(v, m, 32));
      rmax[e] = v;
    }
    if (l16 == 0)
      for (int e = 0; e < 8; ++e) redmax[nQw][mS + e + 8 * half] = rmax[e];
    __syncthreads();

    if (tid < 32) {
      float tm = fmaxf(fmaxf(redmax[0][tid], redmax[1][tid]),
                       fmaxf(redmax[2][tid], redmax[3][tid]));
      float mo = rowm[tid];
      float mn = fmaxf(mo, tm);
      rowm[tid] = mn;
      rsf[tid]  = __expf(mo - mn);                 // first iter -> 0
    }
    __syncthreads();

    // ---- P = exp(S - m), stage to LDS bf16, row sums ----
    float rsum[8];
#pragma unroll
    for (int e = 0; e < 8; ++e) {
      const int r = mS + e + 8 * half;
      float p = __expf(sacc[e] - rowm[r]);
      Ps[r][nS + l16] = (__bf16)p;
      float v = p;
      for (int m = 8; m >= 1; m >>= 1) v += __shfl_xor(v, m, 32);
      rsum[e] = v;
    }
    if (l16 == 0)
      for (int e = 0; e < 8; ++e) redsum[nQw][mS + e + 8 * half] = rsum[e];
    __syncthreads();

    if (tid < 32)
      rowl[tid] = rowl[tid] * rsf[tid] +
                  redsum[0][tid] + redsum[1][tid] + redsum[2][tid] + redsum[3][tid];

    // ---- rescale O, then O += P @ V (K-dim 64 = 2 chunks of 32) ----
#pragma unroll
    for (int mi = 0; mi < 2; ++mi)
      for (int e = 0; e < 8; ++e) {
        const float f = rsf[16 * mi + e + 8 * half];
        for (int ni = 0; ni < 4; ++ni) O[mi][ni][e] *= f;
      }
#pragma unroll
    for (int kc2 = 0; kc2 < 2; ++kc2)
      for (int mi = 0; mi < 2; ++mi) {
        v16bf pa = make_frag2(&Ps[16 * mi + l16][kc2 * 32 + half * 16],
                              &Ps[16 * mi + l16][kc2 * 32 + half * 16 + 8]);
#pragma unroll
        for (int ni = 0; ni < 4; ++ni) {
          const __bf16* vrow = Vb + (size_t)(64 * w + 16 * ni + l16) * S_ +
                               k0 + kc2 * 32 + half * 16;
          O[mi][ni] = wmma_bf16(pa, make_frag16(vrow), O[mi][ni]);
        }
      }
    __syncthreads();
  }

  // ---- normalize and store attn_h [b][s][c] bf16 ----
  __bf16* Ab = Ah + (size_t)b * S_ * C_;
  for (int mi = 0; mi < 2; ++mi)
    for (int e = 0; e < 8; ++e) {
      const int r = 16 * mi + e + 8 * half;
      const float inv = 1.0f / rowl[r];
      for (int ni = 0; ni < 4; ++ni)
        Ab[(size_t)(q0 + r) * C_ + 64 * w + 16 * ni + l16] =
            (__bf16)(O[mi][ni][e] * inv);
    }
}

// ---------------------------------------------------------------------------
// 5) Output projection + bias + residual, written in [b][c][h][w] layout.
//    M = out-channel, N = spatial (so f32 stores coalesce across lanes).
//    No LDS: Wo (0.5MB) and attn_h (33MB) live in L2.
// ---------------------------------------------------------------------------
__global__ __launch_bounds__(256, 1)
void outproj_kernel(const float* __restrict__ x, const __bf16* __restrict__ wob,
                    const float* __restrict__ bo, const __bf16* __restrict__ Ah,
                    float* __restrict__ out) {
  const int b  = blockIdx.z;
  const int o0 = blockIdx.y * 32;
  const int s0 = blockIdx.x * 256;
  const int tid  = threadIdx.x;
  const int w    = tid >> 5;
  const int lane = tid & 31;
  const int l16  = lane & 15;
  const int half = lane >> 4;
  const int mo = 16 * (w >> 2);                    // 0/16 within o-tile
  const int ns = 64 * (w & 3);                     // 0..192 within s-tile

  v8f acc[4];
  for (int ni = 0; ni < 4; ++ni)
    for (int e = 0; e < 8; ++e) acc[ni][e] = 0.f;

  const __bf16* worow = wob + (size_t)(o0 + mo + l16) * C_;
#pragma unroll 4
  for (int kc = 0; kc < C_; kc += 32) {
    v16bf a = make_frag2(&worow[kc + half * 16], &worow[kc + half * 16 + 8]);
#pragma unroll
    for (int ni = 0; ni < 4; ++ni) {
      const __bf16* hrow = Ah + ((size_t)b * S_ + s0 + ns + 16 * ni + l16) * C_ +
                           kc + half * 16;
      acc[ni] = wmma_bf16(a, make_frag16(hrow), acc[ni]);
    }
  }
  for (int ni = 0; ni < 4; ++ni)
    for (int e = 0; e < 8; ++e) {
      const int o = o0 + mo + e + 8 * half;
      const int s = s0 + ns + 16 * ni + l16;
      const size_t idx = ((size_t)b * C_ + o) * S_ + s;
      out[idx] = acc[ni][e] + x[idx] + bo[o];
    }
}

// ---------------------------------------------------------------------------
extern "C" void kernel_launch(void* const* d_in, const int* in_sizes, int n_in,
                              void* d_out, int out_size, void* d_ws, size_t ws_size,
                              hipStream_t stream) {
  (void)in_sizes; (void)n_in; (void)out_size; (void)ws_size;
  const float* x     = (const float*)d_in[0];
  const float* gamma = (const float*)d_in[1];
  const float* beta  = (const float*)d_in[2];
  const float* wq    = (const float*)d_in[3];
  const float* bq    = (const float*)d_in[4];
  const float* wk    = (const float*)d_in[5];
  const float* bk    = (const float*)d_in[6];
  const float* wv    = (const float*)d_in[7];
  const float* bv    = (const float*)d_in[8];
  const float* wo    = (const float*)d_in[9];
  const float* bo    = (const float*)d_in[10];
  float* out = (float*)d_out;

  char* ws = (char*)d_ws;
  float*  stats = (float*)ws;                               // 2 KB
  __bf16* wqb = (__bf16*)(ws + 4096);                       // 4 x 512KB
  __bf16* wkb = wqb + C_ * C_;
  __bf16* wvb = wkb + C_ * C_;
  __bf16* wob = wvb + C_ * C_;
  __bf16* Qg  = (__bf16*)(ws + 4096 + (size_t)4 * C_ * C_ * 2);
  __bf16* Kg  = Qg + (size_t)B_ * S_ * C_;                  // 32 MB each
  __bf16* Vt  = Kg + (size_t)B_ * S_ * C_;
  __bf16* Ah  = Vt + (size_t)B_ * S_ * C_;

  gn_stats_kernel<<<B_ * GRP, 256, 0, stream>>>(x, stats);
  wconv_kernel<<<512, 256, 0, stream>>>(wq, wk, wv, wo, wqb, wkb, wvb, wob);
  qkv_kernel<<<dim3(S_ / 64, C_ / 32, B_), 256, 0, stream>>>(
      x, gamma, beta, stats, wqb, wkb, wvb, bq, bk, bv, Qg, Kg, Vt);
  attn_kernel<<<dim3(S_ / 32, B_), 256, 0, stream>>>(Qg, Kg, Vt, Ah);
  outproj_kernel<<<dim3(S_ / 256, C_ / 32, B_), 256, 0, stream>>>(
      x, wob, bo, Ah, out);
}
